// GATEncoder_34763465294553
// MI455X (gfx1250) — matrix operand order repaired and checked by reading, not compiled, not run on hardware
//
#include <hip/hip_runtime.h>
#include <hip/hip_bf16.h>
#include <stdint.h>

#define HEADS     4
#define HEAD_DIM  32
#define CH        128       // in/out channels, both layers
#define NEG_SLOPE 0.2f

typedef __attribute__((ext_vector_type(16))) _Float16 v16h;
typedef __attribute__((ext_vector_type(8)))  float    v8f;

// ---- order-preserving float <-> uint encoding for atomic segment-max ----
__device__ __forceinline__ unsigned enc_f32(float f) {
  unsigned u = __float_as_uint(f);
  return (u & 0x80000000u) ? ~u : (u | 0x80000000u);
}
__device__ __forceinline__ float dec_f32(unsigned u) {
  return (u & 0x80000000u) ? __uint_as_float(u & 0x7fffffffu)
                           : __uint_as_float(~u);
}
#define ENC_NEG_INF 0x007FFFFFu   // enc(-inf)

// ---------------------------------------------------------------------------
// Pre-pack W (128x128 f32, row-major K x N) into f16 WMMA B-fragments.
// Layout: WB[(((ct*4 + kc)*32 + lane)*16 + e)] = W[(kc*32 + (e>>3)*16 + (lane>>4)*8 + (e&7))*CH
//                                                  + ct*16 + (lane&15)]
// so each lane of each (col-tile, K-chunk) reads one contiguous 32B v16h.
// ---------------------------------------------------------------------------
__global__ void gat_pack_w(const float* __restrict__ W, _Float16* __restrict__ WB)
{
  int i = blockIdx.x * blockDim.x + threadIdx.x;
  if (i >= CH * CH) return;
  const int e    = i & 15;
  const int lane = (i >> 4) & 31;
  const int kc   = (i >> 9) & 3;
  const int ct   = i >> 11;
  const int k = kc * 32 + (e >> 3) * 16 + (lane >> 4) * 8 + (e & 7);
  const int c = ct * 16 + (lane & 15);
  WB[i] = (_Float16)W[(size_t)k * CH + c];
}

// ---------------------------------------------------------------------------
// WMMA GEMM: H[n,128] = f32( f16(relu?(X[n,128])) @ W_f16[128,128] )
// 4 waves/block; each wave computes a 16x32 strip: one shared A fragment feeds
// two v_wmma_f32_16x16x32_f16 per K32 chunk. A loads are 4 x b128 per chunk.
// ---------------------------------------------------------------------------
__device__ __forceinline__ float4 relu4(float4 v) {
  v.x = fmaxf(v.x, 0.f); v.y = fmaxf(v.y, 0.f);
  v.z = fmaxf(v.z, 0.f); v.w = fmaxf(v.w, 0.f);
  return v;
}

template <bool RELU>
__global__ __launch_bounds__(128) void gat_gemm_wmma(
    const float* __restrict__ X, const _Float16* __restrict__ WB,
    float* __restrict__ H, int n_rows)
{
  const int wave = threadIdx.x >> 5;
  const int lane = threadIdx.x & 31;
  const int m    = lane & 15;     // A row / B,D column within tile
  const int hb   = lane >> 4;     // half-wave selector
  const int row0 = blockIdx.x * 16;

  int arow = row0 + m;
  if (arow >= n_rows) arow = n_rows - 1;          // clamp: no predicated loads
  const float* xr = X + (size_t)arow * CH;

  const int ct0 = wave * 2;                       // two column tiles per wave
  v8f acc0 = {}, acc1 = {};

  #pragma unroll
  for (int kc = 0; kc < 4; ++kc) {
    // A fragment (16-bit A 16x32, ISA 7.12.2): per lane, two contiguous runs
    // of 8 K-values: [kc*32 + hb*8, +8) and [kc*32 + 16 + hb*8, +8).
    const float4* p0 = (const float4*)(xr + kc * 32 + hb * 8);
    const float4* p1 = (const float4*)(xr + kc * 32 + 16 + hb * 8);
    float4 x0 = p0[0], x1 = p0[1], x2 = p1[0], x3 = p1[1];
    if (RELU) { x0 = relu4(x0); x1 = relu4(x1); x2 = relu4(x2); x3 = relu4(x3); }

    v16h a;
    a[0]  = (_Float16)x0.x; a[1]  = (_Float16)x0.y;
    a[2]  = (_Float16)x0.z; a[3]  = (_Float16)x0.w;
    a[4]  = (_Float16)x1.x; a[5]  = (_Float16)x1.y;
    a[6]  = (_Float16)x1.z; a[7]  = (_Float16)x1.w;
    a[8]  = (_Float16)x2.x; a[9]  = (_Float16)x2.y;
    a[10] = (_Float16)x2.z; a[11] = (_Float16)x2.w;
    a[12] = (_Float16)x3.x; a[13] = (_Float16)x3.y;
    a[14] = (_Float16)x3.z; a[15] = (_Float16)x3.w;

    // B fragments: contiguous 32B per lane from the pre-packed buffer
    const v16h b0 = *(const v16h*)(WB + ((size_t)(((ct0 + 0) * 4 + kc) * 32 + lane) << 4));
    const v16h b1 = *(const v16h*)(WB + ((size_t)(((ct0 + 1) * 4 + kc) * 32 + lane) << 4));

    acc0 = __builtin_amdgcn_wmma_f32_16x16x32_f16(
        false, a, false, b0, (short)0, acc0, false, false);
    acc1 = __builtin_amdgcn_wmma_f32_16x16x32_f16(
        false, a, false, b1, (short)0, acc1, false, false);
  }

  // D layout: VGPR v holds row M = v + 8*hb, column N = m
  if (row0 + 16 <= n_rows) {
    #pragma unroll
    for (int v = 0; v < 8; ++v) {
      float* hr = H + (size_t)(row0 + v + hb * 8) * CH + ct0 * 16 + m;
      hr[0]  = acc0[v];
      hr[16] = acc1[v];
    }
  } else {
    #pragma unroll
    for (int v = 0; v < 8; ++v) {
      const int r = row0 + v + hb * 8;
      if (r < n_rows) {
        float* hr = H + (size_t)r * CH + ct0 * 16 + m;
        hr[0]  = acc0[v];
        hr[16] = acc1[v];
      }
    }
  }
}

// ---------------------------------------------------------------------------
// alpha_s / alpha_d: per (node, head) 32-wide dot products
// ---------------------------------------------------------------------------
__global__ void gat_alpha(const float* __restrict__ H,
                          const float* __restrict__ a_src,
                          const float* __restrict__ a_dst,
                          float* __restrict__ AS, float* __restrict__ AD, int n)
{
  int i = blockIdx.x * blockDim.x + threadIdx.x;
  if (i >= n * HEADS) return;
  const int node = i >> 2, hd = i & (HEADS - 1);
  const float* hp = H + (size_t)node * CH + hd * HEAD_DIM;
  float s = 0.f, d = 0.f;
  #pragma unroll
  for (int c = 0; c < HEAD_DIM; ++c) {
    const float v = hp[c];
    s = fmaf(v, a_src[hd * HEAD_DIM + c], s);
    d = fmaf(v, a_dst[hd * HEAD_DIM + c], d);
  }
  AS[i] = s; AD[i] = d;
}

// ---------------------------------------------------------------------------
// Per-node init: AGG = bias broadcast (reference adds b AFTER segment-sum),
// EMAX = enc(-inf), DEN = 0
// ---------------------------------------------------------------------------
__global__ void gat_init_node(unsigned* __restrict__ EMAX, float* __restrict__ DEN,
                              float* __restrict__ AGG, const float* __restrict__ bias,
                              int n)
{
  int i = blockIdx.x * blockDim.x + threadIdx.x;
  if (i >= n * CH) return;
  const int ch = i & (CH - 1);
  AGG[i] = bias[ch];
  if (ch < HEADS) {
    const int node = i >> 7;
    EMAX[node * HEADS + ch] = ENC_NEG_INF;
    DEN[node * HEADS + ch]  = 0.0f;
  }
}

__device__ __forceinline__ void edge_sd(const long long* __restrict__ ei,
                                        int e, int n_edges, int& s, int& d)
{
  if (e < n_edges) { s = (int)ei[e]; d = (int)ei[n_edges + e]; }
  else             { s = d = e - n_edges; }   // self-loops appended
}

// ---------------------------------------------------------------------------
// Pass 1 over edges: e = lrelu(as[src] + ad[dst]); stash raw e; segment max
// ---------------------------------------------------------------------------
__global__ void gat_edge_lrelu_max(const long long* __restrict__ ei,
                                   const float* __restrict__ AS,
                                   const float* __restrict__ AD,
                                   float* __restrict__ EE,
                                   unsigned* __restrict__ EMAX,
                                   int n_edges, int e_tot)
{
  int i = blockIdx.x * blockDim.x + threadIdx.x;
  if (i >= e_tot * HEADS) return;
  const int e = i >> 2, hd = i & (HEADS - 1);
  int s, d; edge_sd(ei, e, n_edges, s, d);
  float v = AS[s * HEADS + hd] + AD[d * HEADS + hd];
  v = (v > 0.f) ? v : NEG_SLOPE * v;
  EE[i] = v;
  atomicMax(&EMAX[d * HEADS + hd], enc_f32(v));
}

// ---------------------------------------------------------------------------
// Pass 2: ee = exp(e - max[dst]); segment-sum into DEN
// ---------------------------------------------------------------------------
__global__ void gat_edge_exp_sum(const long long* __restrict__ ei,
                                 float* __restrict__ EE,
                                 const unsigned* __restrict__ EMAX,
                                 float* __restrict__ DEN,
                                 int n_edges, int e_tot)
{
  int i = blockIdx.x * blockDim.x + threadIdx.x;
  if (i >= e_tot * HEADS) return;
  const int e = i >> 2, hd = i & (HEADS - 1);
  int s, d; edge_sd(ei, e, n_edges, s, d);
  const float v = __expf(EE[i] - dec_f32(EMAX[d * HEADS + hd]));
  EE[i] = v;
  atomicAdd(&DEN[d * HEADS + hd], v);
}

// ---------------------------------------------------------------------------
// Pass 3: weighted scatter: AGG[dst, ch] += h[src, ch] * ee/(den+eps)
// One 128-thread block per edge (thread = channel, head = ch/32).
// ---------------------------------------------------------------------------
__global__ __launch_bounds__(CH) void gat_edge_scatter(
    const long long* __restrict__ ei,
    const float* __restrict__ H, const float* __restrict__ EE,
    const float* __restrict__ DEN, float* __restrict__ AGG, int n_edges)
{
  const int e  = blockIdx.x;
  const int ch = threadIdx.x;
  const int hd = ch >> 5;
  int s, d; edge_sd(ei, e, n_edges, s, d);
  const float alpha = EE[(size_t)e * HEADS + hd] /
                      (DEN[d * HEADS + hd] + 1e-16f);
  atomicAdd(&AGG[(size_t)d * CH + ch], H[(size_t)s * CH + ch] * alpha);
}

// ---------------------------------------------------------------------------
extern "C" void kernel_launch(void* const* d_in, const int* in_sizes, int n_in,
                              void* d_out, int out_size, void* d_ws, size_t ws_size,
                              hipStream_t stream)
{
  (void)n_in; (void)out_size; (void)ws_size;

  const float*     x   = (const float*)d_in[0];
  const long long* ei  = (const long long*)d_in[1];  // int64 [2, E]
  const float*     W1  = (const float*)d_in[2];
  const float*     as1 = (const float*)d_in[3];
  const float*     ad1 = (const float*)d_in[4];
  const float*     b1  = (const float*)d_in[5];
  const float*     W2  = (const float*)d_in[6];
  const float*     as2 = (const float*)d_in[7];
  const float*     ad2 = (const float*)d_in[8];
  const float*     b2  = (const float*)d_in[9];

  const int n_nodes = in_sizes[0] / CH;   // 50000
  const int n_edges = in_sizes[1] / 2;    // 640000
  const int e_tot   = n_edges + n_nodes;  // + self loops

  // ---- carve workspace (all buffers fully (re)initialized every call) ----
  float* ws = (float*)d_ws;
  size_t o = 0;
  _Float16* WB1 = (_Float16*)(ws + o);   o += (CH * CH) / 2;   // f16 packed W1
  _Float16* WB2 = (_Float16*)(ws + o);   o += (CH * CH) / 2;   // f16 packed W2
  float*    H    = ws + o;               o += (size_t)n_nodes * CH;
  float*    AS   = ws + o;               o += (size_t)n_nodes * HEADS;
  float*    AD   = ws + o;               o += (size_t)n_nodes * HEADS;
  unsigned* EMAX = (unsigned*)(ws + o);  o += (size_t)n_nodes * HEADS;
  float*    DEN  = ws + o;               o += (size_t)n_nodes * HEADS;
  float*    EE   = ws + o;               o += (size_t)e_tot  * HEADS;
  float*    AGG1 = ws + o;               o += (size_t)n_nodes * CH;
  float*    OUT  = (float*)d_out;

  const int TPB    = 256;
  const int gAlpha = (n_nodes * HEADS + TPB - 1) / TPB;
  const int gInit  = (n_nodes * CH    + TPB - 1) / TPB;
  const int gEdge  = (e_tot   * HEADS + TPB - 1) / TPB;
  const int gGemm  = (n_nodes + 15) / 16;
  const int gPack  = (CH * CH + TPB - 1) / TPB;

  // ---- pre-pack both weight matrices into WMMA B-fragment layout ----
  gat_pack_w<<<gPack, TPB, 0, stream>>>(W1, WB1);
  gat_pack_w<<<gPack, TPB, 0, stream>>>(W2, WB2);

  // ---------------- layer 1 ----------------
  gat_gemm_wmma<false><<<gGemm, 128, 0, stream>>>(x, WB1, H, n_nodes);
  gat_alpha    <<<gAlpha, TPB, 0, stream>>>(H, as1, ad1, AS, AD, n_nodes);
  gat_init_node<<<gInit,  TPB, 0, stream>>>(EMAX, DEN, AGG1, b1, n_nodes);
  gat_edge_lrelu_max<<<gEdge, TPB, 0, stream>>>(ei, AS, AD, EE, EMAX, n_edges, e_tot);
  gat_edge_exp_sum  <<<gEdge, TPB, 0, stream>>>(ei, EE, EMAX, DEN, n_edges, e_tot);
  gat_edge_scatter  <<<e_tot, CH,  0, stream>>>(ei, H, EE, DEN, AGG1, n_edges);

  // ---------------- layer 2 (ReLU fused into GEMM A-load) ----------------
  gat_gemm_wmma<true><<<gGemm, 128, 0, stream>>>(AGG1, WB2, H, n_nodes);
  gat_alpha    <<<gAlpha, TPB, 0, stream>>>(H, as2, ad2, AS, AD, n_nodes);
  gat_init_node<<<gInit,  TPB, 0, stream>>>(EMAX, DEN, OUT, b2, n_nodes);
  gat_edge_lrelu_max<<<gEdge, TPB, 0, stream>>>(ei, AS, AD, EE, EMAX, n_edges, e_tot);
  gat_edge_exp_sum  <<<gEdge, TPB, 0, stream>>>(ei, EE, EMAX, DEN, n_edges, e_tot);
  gat_edge_scatter  <<<e_tot, CH,  0, stream>>>(ei, H, EE, DEN, OUT, n_edges);
}
